// Linear_85959475462525
// MI455X (gfx1250) — compile-verified, matrix operand order checked by reading, not run on hardware
//
#include <hip/hip_runtime.h>
#include <hip/hip_bf16.h>

// Problem dimensions (fixed by the reference).
#define M_TOT 8192   // B*S = 4*2048
#define N_TOT 4096   // D_OUT
#define K_TOT 4096   // D_IN

// GEMM tiling.
#define BM 128
#define BN 128
#define BK 32
#define LSTR 40      // LDS row stride in bf16 elements (32 + 8 pad -> conflict-free, 16B aligned)

typedef __attribute__((ext_vector_type(16))) __bf16 v16bf;
typedef __attribute__((ext_vector_type(8)))  __bf16 v8bf;
typedef __attribute__((ext_vector_type(4)))  __bf16 v4bf;
typedef __attribute__((ext_vector_type(8)))  float  v8f;

// Load a 16x32-layout WMMA fragment (16 bf16 halves per lane) from LDS.
// Halves [0..7]  = K in [koff,   koff+7]
// Halves [8..15] = K in [koff+16,koff+23]   (per CDNA5 16-bit A/B VGPR layout)
static __device__ inline v16bf ldfrag(const __bf16* p) {
    v8bf a = *(const v8bf*)p;
    v8bf b = *(const v8bf*)(p + 16);
    return __builtin_shufflevector(a, b, 0, 1, 2, 3, 4, 5, 6, 7,
                                         8, 9, 10, 11, 12, 13, 14, 15);
}

__global__ __launch_bounds__(256)
void gemm_kernel(const float* __restrict__ X,     // [M_TOT, K_TOT]
                 const float* __restrict__ W,     // [N_TOT, K_TOT]
                 const float* __restrict__ bias,  // [N_TOT]
                 const float* __restrict__ ood,   // [2]
                 float* __restrict__ out)         // [M_TOT, N_TOT]
{
    // Split-precision bf16 tiles: hi + lo for both operands (40 KB total).
    __shared__ __bf16 Ah[BM * LSTR];
    __shared__ __bf16 Al[BM * LSTR];
    __shared__ __bf16 Bh[BN * LSTR];
    __shared__ __bf16 Bl[BN * LSTR];

    const int t    = threadIdx.x;
    const int m0   = blockIdx.y * BM;
    const int n0   = blockIdx.x * BN;

    const int lane = t & 31;
    const int wave = t >> 5;
    const int wm   = wave & 1;        // wave M block: 0..1 (64 rows each)
    const int wn   = wave >> 1;       // wave N block: 0..3 (32 cols each)
    const int lr   = lane & 15;
    const int half = lane >> 4;       // 0: lanes 0-15, 1: lanes 16-31
    const int koff = half * 8;

    // Per-thread staging coordinates: 128 rows x 8 float4-chunks = 1024 chunks, 4/thread.
    int srow[4], scol[4];
    #pragma unroll
    for (int j = 0; j < 4; ++j) {
        const int chunk = t + 256 * j;
        srow[j] = chunk >> 3;
        scol[j] = (chunk & 7) * 4;
    }

    v8f acc[4][2] = {};

    // ---- Software pipeline: preload K-step 0 into registers --------------------
    float4 xa[4], wb[4];
    #pragma unroll
    for (int j = 0; j < 4; ++j) {
        xa[j] = *(const float4*)(X + (size_t)(m0 + srow[j]) * K_TOT + scol[j]);
        wb[j] = *(const float4*)(W + (size_t)(n0 + srow[j]) * K_TOT + scol[j]);
    }

    for (int k0 = 0; k0 < K_TOT; k0 += BK) {
        // ---- Convert staged fp32 regs -> bf16(hi,lo) tiles in LDS --------------
        #pragma unroll
        for (int j = 0; j < 4; ++j) {
            float xs[4] = {xa[j].x, xa[j].y, xa[j].z, xa[j].w};
            float ws[4] = {wb[j].x, wb[j].y, wb[j].z, wb[j].w};
            v4bf ah, al, bh, bl;
            #pragma unroll
            for (int e = 0; e < 4; ++e) {
                __bf16 h = (__bf16)xs[e];
                ah[e] = h;
                al[e] = (__bf16)(xs[e] - (float)h);
                __bf16 g = (__bf16)ws[e];
                bh[e] = g;
                bl[e] = (__bf16)(ws[e] - (float)g);
            }
            *(v4bf*)&Ah[srow[j] * LSTR + scol[j]] = ah;
            *(v4bf*)&Al[srow[j] * LSTR + scol[j]] = al;
            *(v4bf*)&Bh[srow[j] * LSTR + scol[j]] = bh;
            *(v4bf*)&Bl[srow[j] * LSTR + scol[j]] = bl;
        }
        __syncthreads();

        // ---- Issue next K-step's global loads; they overlap the WMMA phase -----
        const int kn = k0 + BK;
        if (kn < K_TOT) {
            #pragma unroll
            for (int j = 0; j < 4; ++j) {
                xa[j] = *(const float4*)(X + (size_t)(m0 + srow[j]) * K_TOT + kn + scol[j]);
                wb[j] = *(const float4*)(W + (size_t)(n0 + srow[j]) * K_TOT + kn + scol[j]);
            }
            // Warm L2 two K-steps ahead (global_prefetch_b8).
            const int kp = k0 + 2 * BK;
            if (kp < K_TOT) {
                __builtin_prefetch(X + (size_t)(m0 + srow[0]) * K_TOT + kp + scol[0], 0, 1);
                __builtin_prefetch(W + (size_t)(n0 + srow[0]) * K_TOT + kp + scol[0], 0, 1);
            }
        }

        // ---- Fragment loads (ds_load_b128 pairs) -------------------------------
        v16bf afh[4], afl[4], bfh[2], bfl[2];
        #pragma unroll
        for (int tm = 0; tm < 4; ++tm) {
            const int r = (wm * 64 + tm * 16 + lr) * LSTR + koff;
            afh[tm] = ldfrag(&Ah[r]);
            afl[tm] = ldfrag(&Al[r]);
        }
        #pragma unroll
        for (int tn = 0; tn < 2; ++tn) {
            const int r = (wn * 32 + tn * 16 + lr) * LSTR + koff;
            bfh[tn] = ldfrag(&Bh[r]);
            bfl[tn] = ldfrag(&Bl[r]);
        }

        // ---- 3-product split-precision WMMA accumulate -------------------------
        #pragma unroll
        for (int tm = 0; tm < 4; ++tm) {
            #pragma unroll
            for (int tn = 0; tn < 2; ++tn) {
                acc[tm][tn] = __builtin_amdgcn_wmma_f32_16x16x32_bf16(
                    false, afh[tm], false, bfh[tn], (short)0, acc[tm][tn], false, false);
                acc[tm][tn] = __builtin_amdgcn_wmma_f32_16x16x32_bf16(
                    false, afh[tm], false, bfl[tn], (short)0, acc[tm][tn], false, false);
                acc[tm][tn] = __builtin_amdgcn_wmma_f32_16x16x32_bf16(
                    false, afl[tm], false, bfh[tn], (short)0, acc[tm][tn], false, false);
            }
        }
        __syncthreads();
    }

    // ---- Epilogue: out = w0 * (acc + bias) ------------------------------------
    const float w0 = ood[0];
    #pragma unroll
    for (int tm = 0; tm < 4; ++tm) {
        #pragma unroll
        for (int tn = 0; tn < 2; ++tn) {
            const int nn = n0 + wn * 32 + tn * 16 + lr;
            const float bn = bias[nn];
            const int mb = m0 + wm * 64 + tm * 16 + half * 8;
            #pragma unroll
            for (int v = 0; v < 8; ++v) {
                out[(size_t)(mb + v) * N_TOT + nn] = w0 * (acc[tm][tn][v] + bn);
            }
        }
    }
}

// Enforce `.set` last-write-wins semantics for duplicate (row,col) pairs.
__global__ void dedupe_kernel(const int* __restrict__ row, const int* __restrict__ col,
                              unsigned char* __restrict__ flags, int nf)
{
    const int f = blockIdx.x * blockDim.x + threadIdx.x;
    if (f >= nf) return;
    const int r = row[f], c = col[f];
    unsigned char ok = 1;
    for (int g = f + 1; g < nf; ++g) {
        if (row[g] == r && col[g] == c) { ok = 0; break; }
    }
    flags[f] = ok;
}

// Sparse FourierFT term: out[m, row[f]] += w1 * spectrum[f] * x[m, col[f]]
__global__ __launch_bounds__(256)
void scatter_kernel(const float* __restrict__ x, const float* __restrict__ spectrum,
                    const int* __restrict__ row, const int* __restrict__ col,
                    const unsigned char* __restrict__ flags, const float* __restrict__ ood,
                    float* __restrict__ out, int nf)
{
    const int m = blockIdx.x;
    const float w1 = ood[1];
    const float* xr = x + (size_t)m * K_TOT;
    float* orow = out + (size_t)m * N_TOT;
    for (int f = threadIdx.x; f < nf; f += blockDim.x) {
        if (!flags[f]) continue;
        const float v = w1 * spectrum[f] * xr[col[f]];
        atomicAdd(&orow[row[f]], v);
    }
}

// o_loss = sum_k ( dot(o_spectra[k], spectrum) )^2   (K = 2)
__global__ void oloss_kernel(const float* __restrict__ spectrum,
                             const float* __restrict__ o_spectra,
                             float* __restrict__ out_loss, int nf)
{
    __shared__ float s0[256];
    __shared__ float s1[256];
    float d0 = 0.f, d1 = 0.f;
    for (int f = threadIdx.x; f < nf; f += blockDim.x) {
        const float s = spectrum[f];
        d0 += o_spectra[f] * s;
        d1 += o_spectra[nf + f] * s;
    }
    s0[threadIdx.x] = d0;
    s1[threadIdx.x] = d1;
    __syncthreads();
    if (threadIdx.x == 0) {
        float a = 0.f, b = 0.f;
        for (int i = 0; i < 256; ++i) { a += s0[i]; b += s1[i]; }
        out_loss[0] = a * a + b * b;
    }
}

extern "C" void kernel_launch(void* const* d_in, const int* in_sizes, int n_in,
                              void* d_out, int out_size, void* d_ws, size_t ws_size,
                              hipStream_t stream)
{
    const float* x        = (const float*)d_in[0];
    const float* weight   = (const float*)d_in[1];
    const float* bias     = (const float*)d_in[2];
    const float* spectrum = (const float*)d_in[3];
    const float* ood      = (const float*)d_in[4];
    const float* o_spec   = (const float*)d_in[5];
    const int*   row_idx  = (const int*)d_in[6];
    const int*   col_idx  = (const int*)d_in[7];
    float*       out      = (float*)d_out;
    unsigned char* flags  = (unsigned char*)d_ws;
    const int nf          = in_sizes[3];   // 1000

    // 1) Fused GEMM + bias + w0 scale (writes every element of out[0 .. M*N)).
    gemm_kernel<<<dim3(N_TOT / BN, M_TOT / BM), 256, 0, stream>>>(x, weight, bias, ood, out);

    // 2) Duplicate-index resolution (.set semantics).
    dedupe_kernel<<<(nf + 255) / 256, 256, 0, stream>>>(row_idx, col_idx, flags, nf);

    // 3) Sparse delta-weight contribution, scaled by w1 (atomic adds on top of out).
    scatter_kernel<<<M_TOT, 256, 0, stream>>>(x, spectrum, row_idx, col_idx, flags, ood, out, nf);

    // 4) Orthogonality loss scalar at out[M*N].
    oloss_kernel<<<1, 256, 0, stream>>>(spectrum, o_spec, out + (size_t)M_TOT * N_TOT, nf);
}
